// MultiNatten_52475910422719
// MI455X (gfx1250) — compile-verified, matrix operand order checked by reading, not run on hardware
//
#include <hip/hip_runtime.h>
#include <hip/hip_bf16.h>
#include <math.h>

typedef __attribute__((ext_vector_type(16))) _Float16 v16h;
typedef __attribute__((ext_vector_type(8)))  float    v8f;

#define HGT   56
#define WID   56
#define BATCH 4
#define NPIX  (BATCH*HGT*WID)   /* 12544 tokens per 64-ch branch */
#define NPOOL (BATCH*28*28)     /* 3136 pooled tokens (global attn) */

// ---------------------------------------------------------------------------
// f32 [K,N] -> f16 [N,K] (transpose) weight convert. Reads coalesced; the
// transposed layout makes B-fragment loads contiguous per lane so the GEMM
// can use global_load_b128 for both A and B fragments.
// ---------------------------------------------------------------------------
__global__ __launch_bounds__(256) void k_cvtT16(const float* __restrict__ s,
                                                _Float16* __restrict__ d,
                                                int K, int N) {
  int i = blockIdx.x * 256 + threadIdx.x;
  if (i >= K * N) return;
  int k = i / N, n = i - k * N;
  d[(long)n * K + k] = (_Float16)s[i];
}

// ---------------------------------------------------------------------------
// LayerNorm over 64 channels; one wave32 per row, 2 elements per lane.
// Reads strided (ldx) so it can slice a 64-ch part out of the 256-ch input.
// ---------------------------------------------------------------------------
template<bool W16, bool W32>
__global__ __launch_bounds__(128) void k_ln64(
    const float* __restrict__ X, int ldx,
    const float* __restrict__ g, const float* __restrict__ b,
    _Float16* __restrict__ o16, float* __restrict__ o32, int M)
{
  int row = blockIdx.x * 4 + (threadIdx.x >> 5);
  if (row >= M) return;
  int lane = threadIdx.x & 31;
  const float* xp = X + (long)row * ldx;
  float a0 = xp[lane], a1 = xp[lane + 32];
  float s = a0 + a1;
  #pragma unroll
  for (int m = 16; m; m >>= 1) s += __shfl_xor(s, m, 32);
  float mean = s * (1.0f / 64.0f);
  float d0 = a0 - mean, d1 = a1 - mean;
  float v = d0 * d0 + d1 * d1;
  #pragma unroll
  for (int m = 16; m; m >>= 1) v += __shfl_xor(v, m, 32);
  float inv = rsqrtf(v * (1.0f / 64.0f) + 1e-5f);
  float y0 = d0 * inv * g[lane]      + b[lane];
  float y1 = d1 * inv * g[lane + 32] + b[lane + 32];
  long o = (long)row * 64;
  if (W16) { o16[o + lane] = (_Float16)y0; o16[o + lane + 32] = (_Float16)y1; }
  if (W32) { o32[o + lane] = y0;           o32[o + lane + 32] = y1; }
}

// ---------------------------------------------------------------------------
// WMMA GEMM: C[M,N] = A[M,K](f16, row-major) * Bt[N,K](f16, weights stored
// transposed) + bias. One wave per 16x16 tile, K stepped by 32 via
// v_wmma_f32_16x16x32_f16 (f32 accumulate).
//
// Fragment layouts per CDNA5 ISA 7.12.2 (wave32):
//   A: lane -> row = lane%16 ; pair j holds K = (j/4)*16 + (lane/16)*8 + (j&3)*2
//   B: lane -> col = lane%16 ; same K pattern  (contiguous in Bt[N,K])
//   C/D: vgpr r -> row = r + (lane/16)*8, col = lane%16
// Both fragments therefore load as 2x global_load_b128 per k-step.
//
// Epilogue fuses: exact GELU (ACT=1), residual add (strided), f32 and/or f16
// stores with independent leading dims (so fc2 writes straight into d_out).
// ---------------------------------------------------------------------------
template<int ACT, bool RESID, bool W32, bool W16>
__global__ __launch_bounds__(32) void k_gemm(
    const _Float16* __restrict__ A, int lda,
    const _Float16* __restrict__ Bt, int ldb,   // ldb = K stride of Bt[N,K]
    const float* __restrict__ bias,
    const float* __restrict__ R, int ldr,
    float* __restrict__ C32, int ldc32,
    _Float16* __restrict__ C16, int ldc16,
    int Kd)
{
  const int tn   = blockIdx.x * 16;
  const int tm   = blockIdx.y * 16;
  const int lane = threadIdx.x & 31;
  const int r15  = lane & 15;
  const int hi8  = (lane >> 4) * 8;

  const _Float16* Arow = A  + (long)(tm + r15) * lda;
  const _Float16* Brow = Bt + (long)(tn + r15) * ldb;

  v8f acc = {};
  for (int k0 = 0; k0 < Kd; k0 += 32) {
    v16h af = {}, bf = {};
    #pragma unroll
    for (int j = 0; j < 8; ++j) {
      int k = k0 + ((j >> 2) << 4) + hi8 + ((j & 3) << 1);
      af[2*j]   = Arow[k];
      af[2*j+1] = Arow[k + 1];
      bf[2*j]   = Brow[k];
      bf[2*j+1] = Brow[k + 1];
    }
    acc = __builtin_amdgcn_wmma_f32_16x16x32_f16(false, af, false, bf,
                                                 (short)0, acc, false, false);
  }

  const int col = tn + r15;
  const float bv = bias ? bias[col] : 0.0f;
  #pragma unroll
  for (int r = 0; r < 8; ++r) {
    int row = tm + r + hi8;
    float v = acc[r] + bv;
    if (ACT == 1) v = 0.5f * v * (1.0f + erff(v * 0.70710678118f));
    if (RESID) v += R[(long)row * ldr + col];
    if (W32) C32[(long)row * ldc32 + col] = v;
    if (W16) C16[(long)row * ldc16 + col] = (_Float16)v;
  }
}

// ---------------------------------------------------------------------------
// NATTEN 7x7 clamped-window attention. qkv:[NPIX,192] f32 laid out
// [3,heads=4,d=16] per token. One thread per (token, head): 49 dot16 scores
// + rpb bias, softmax, 49 weighted v16 accumulates. out: f16 [NPIX,64],
// channel = h*16+d (matches reference transpose/reshape).
// ---------------------------------------------------------------------------
__global__ __launch_bounds__(256) void k_natattn(
    const float* __restrict__ qkv, const float* __restrict__ rpb,
    _Float16* __restrict__ out)
{
  int t = blockIdx.x * 256 + threadIdx.x;
  if (t >= NPIX * 4) return;
  const int h   = t & 3;
  const int pix = t >> 2;
  const int xx  = pix % WID;
  const int t2  = pix / WID;
  const int yy  = t2 % HGT;
  const int b   = t2 / HGT;

  float q[16];
  const float* qp = qkv + (long)pix * 192 + h * 16;
  #pragma unroll
  for (int i = 0; i < 16; ++i) q[i] = qp[i] * 0.25f;   // d^-0.5, d=16

  int sy = yy - 3; sy = sy < 0 ? 0 : (sy > HGT - 7 ? HGT - 7 : sy);
  int sx = xx - 3; sx = sx < 0 ? 0 : (sx > WID - 7 ? WID - 7 : sx);
  const float* rp = rpb + h * 13 * 13;

  float s[49];
  float mx = -1e30f;
  for (int a = 0; a < 7; ++a) {
    int ky = sy + a;
    for (int c = 0; c < 7; ++c) {
      int kx = sx + c;
      const float* kp = qkv + (long)((b * HGT + ky) * WID + kx) * 192 + 64 + h * 16;
      float d = 0.0f;
      #pragma unroll
      for (int i = 0; i < 16; ++i) d += q[i] * kp[i];
      d += rp[(ky - yy + 6) * 13 + (kx - xx + 6)];
      s[a * 7 + c] = d;
      mx = fmaxf(mx, d);
    }
  }
  float den = 0.0f;
  #pragma unroll
  for (int i = 0; i < 49; ++i) { s[i] = __expf(s[i] - mx); den += s[i]; }
  const float rden = 1.0f / den;

  float accv[16];
  #pragma unroll
  for (int i = 0; i < 16; ++i) accv[i] = 0.0f;
  for (int a = 0; a < 7; ++a) {
    int ky = sy + a;
    for (int c = 0; c < 7; ++c) {
      int kx = sx + c;
      const float* vp = qkv + (long)((b * HGT + ky) * WID + kx) * 192 + 128 + h * 16;
      float w = s[a * 7 + c] * rden;
      #pragma unroll
      for (int i = 0; i < 16; ++i) accv[i] += w * vp[i];
    }
  }
  _Float16* op = out + (long)pix * 64 + h * 16;
  #pragma unroll
  for (int i = 0; i < 16; ++i) op[i] = (_Float16)accv[i];
}

// ---------------------------------------------------------------------------
// Global-attention helpers: 2x2 average pool (post-LN) and flash-style full
// attention over N=784 pooled tokens, then nearest-upsample + residual add.
// ---------------------------------------------------------------------------
__global__ __launch_bounds__(256) void k_ga_pool(const float* __restrict__ xn,
                                                 _Float16* __restrict__ xa)
{
  int i = blockIdx.x * 256 + threadIdx.x;
  if (i >= NPOOL * 64) return;
  int c  = i & 63;
  int t2 = i >> 6;
  int p  = t2 % 784, b = t2 / 784;
  int py = p / 28, px = p % 28;
  long r0 = (long)(b * HGT + 2 * py) * WID + 2 * px;
  float v = xn[r0 * 64 + c] + xn[(r0 + 1) * 64 + c] +
            xn[(r0 + WID) * 64 + c] + xn[(r0 + WID + 1) * 64 + c];
  xa[i] = (_Float16)(v * 0.25f);
}

__global__ __launch_bounds__(256) void k_ga_attn(const float* __restrict__ qkv,
                                                 float* __restrict__ ao)
{
  int t = blockIdx.x * 256 + threadIdx.x;
  if (t >= NPOOL * 4) return;
  const int h   = t & 3;
  const int idx = t >> 2;          // b*784 + n
  const int b   = idx / 784;

  float q[16];
  const float* qp = qkv + (long)idx * 192 + h * 16;
  #pragma unroll
  for (int i = 0; i < 16; ++i) q[i] = qp[i] * 0.25f;

  const float* kbase = qkv + (long)b * 784 * 192 + 64 + h * 16;
  const float* vbase = qkv + (long)b * 784 * 192 + 128 + h * 16;

  float m = -1e30f, l = 0.0f;
  float accv[16];
  #pragma unroll
  for (int i = 0; i < 16; ++i) accv[i] = 0.0f;

  for (int j = 0; j < 784; ++j) {
    const float* kp = kbase + (long)j * 192;
    float d = 0.0f;
    #pragma unroll
    for (int i = 0; i < 16; ++i) d += q[i] * kp[i];
    float mn   = fmaxf(m, d);
    float corr = __expf(m - mn);
    float w    = __expf(d - mn);
    l = l * corr + w;
    const float* vp = vbase + (long)j * 192;
    #pragma unroll
    for (int i = 0; i < 16; ++i) accv[i] = accv[i] * corr + w * vp[i];
    m = mn;
  }
  float rl = 1.0f / l;
  float* op = ao + (long)idx * 64 + h * 16;
  #pragma unroll
  for (int i = 0; i < 16; ++i) op[i] = accv[i] * rl;
}

__global__ __launch_bounds__(256) void k_ga_upadd(const float* __restrict__ x,
                                                  const float* __restrict__ ao,
                                                  float* __restrict__ x1)
{
  int i = blockIdx.x * 256 + threadIdx.x;
  if (i >= NPIX * 64) return;
  int c   = i & 63;
  int pix = i >> 6;
  int xx  = pix % WID;
  int t2  = pix / WID;
  int yy  = t2 % HGT;
  int b   = t2 / HGT;
  int p   = (yy >> 1) * 28 + (xx >> 1);
  x1[i] = x[(long)pix * 256 + 192 + c] + ao[((long)b * 784 + p) * 64 + c];
}

// ---------------------------------------------------------------------------
// Orchestration
// ---------------------------------------------------------------------------
extern "C" void kernel_launch(void* const* d_in, const int* in_sizes, int n_in,
                              void* d_out, int out_size, void* d_ws, size_t ws_size,
                              hipStream_t stream) {
  const float* x          = (const float*)d_in[0];
  const float* nat_ln1_g  = (const float*)d_in[1];
  const float* nat_ln1_b  = (const float*)d_in[2];
  const float* nat_qkv_w  = (const float*)d_in[3];
  const float* nat_qkv_b  = (const float*)d_in[4];
  const float* nat_rpb    = (const float*)d_in[5];
  const float* nat_proj_w = (const float*)d_in[6];
  const float* nat_proj_b = (const float*)d_in[7];
  const float* nat_ln2_g  = (const float*)d_in[8];
  const float* nat_ln2_b  = (const float*)d_in[9];
  const float* nat_fc1_w  = (const float*)d_in[10];
  const float* nat_fc1_b  = (const float*)d_in[11];
  const float* nat_fc2_w  = (const float*)d_in[12];
  const float* nat_fc2_b  = (const float*)d_in[13];
  const float* ga_ln1_g   = (const float*)d_in[14];
  const float* ga_ln1_b   = (const float*)d_in[15];
  const float* ga_qkv_w   = (const float*)d_in[16];
  const float* ga_qkv_b   = (const float*)d_in[17];
  const float* ga_ln2_g   = (const float*)d_in[18];
  const float* ga_ln2_b   = (const float*)d_in[19];
  const float* ga_fc1_w   = (const float*)d_in[20];
  const float* ga_fc1_b   = (const float*)d_in[21];
  const float* ga_fc2_w   = (const float*)d_in[22];
  const float* ga_fc2_b   = (const float*)d_in[23];
  float* out = (float*)d_out;

  char* ws = (char*)d_ws;
  size_t off = 0;
  auto alloc = [&](size_t bytes) -> void* {
    void* p = ws + off;
    off = (off + bytes + 255) & ~(size_t)255;
    return p;
  };
  float*    qkv32 = (float*)alloc((size_t)NPIX * 192 * 4);  // qkv activations
  float*    x1    = (float*)alloc((size_t)NPIX * 64 * 4);   // post-attn residual
  float*    xn32  = (float*)alloc((size_t)NPIX * 64 * 4);   // GA ln1 (f32, pre-pool)
  _Float16* xn16  = (_Float16*)alloc((size_t)NPIX * 64 * 2);
  _Float16* at16  = (_Float16*)alloc((size_t)NPIX * 64 * 2); // also GA ao (f32, 3136*64 fits)
  _Float16* ln216 = (_Float16*)alloc((size_t)NPIX * 64 * 2);
  _Float16* h116  = (_Float16*)alloc((size_t)NPIX * 256 * 2);
  _Float16* wq16  = (_Float16*)alloc((size_t)64 * 192 * 2);  // Bt layouts [N,K]
  _Float16* wp16  = (_Float16*)alloc((size_t)64 * 64 * 2);
  _Float16* w116  = (_Float16*)alloc((size_t)64 * 256 * 2);
  _Float16* w216  = (_Float16*)alloc((size_t)256 * 64 * 2);

  const int lnGrid = (NPIX + 3) / 4;

  // ---- three NAT branches (buffers reused; kernels serialize on stream) ----
  for (int i = 0; i < 3; ++i) {
    k_cvtT16<<<(64*192 + 255) / 256, 256, 0, stream>>>(nat_qkv_w  + (long)i*64*192, wq16, 64, 192);
    k_cvtT16<<<(64*64  + 255) / 256, 256, 0, stream>>>(nat_proj_w + (long)i*64*64,  wp16, 64, 64);
    k_cvtT16<<<(64*256 + 255) / 256, 256, 0, stream>>>(nat_fc1_w  + (long)i*64*256, w116, 64, 256);
    k_cvtT16<<<(256*64 + 255) / 256, 256, 0, stream>>>(nat_fc2_w  + (long)i*256*64, w216, 256, 64);

    // LN1 (strided slice of x) -> f16
    k_ln64<true, false><<<lnGrid, 128, 0, stream>>>(x + i*64, 256,
        nat_ln1_g + i*64, nat_ln1_b + i*64, xn16, nullptr, NPIX);

    // QKV: [12544,64] x [64,192] + b -> f32
    k_gemm<0, false, true, false><<<dim3(192/16, NPIX/16), 32, 0, stream>>>(
        xn16, 64, wq16, 64, nat_qkv_b + i*192,
        nullptr, 0, qkv32, 192, nullptr, 0, 64);

    // 7x7 neighborhood attention -> f16
    k_natattn<<<(NPIX*4 + 255) / 256, 256, 0, stream>>>(
        qkv32, nat_rpb + (long)i*4*13*13, at16);

    // proj + bias + residual(x slice) -> x1 (f32)
    k_gemm<0, true, true, false><<<dim3(64/16, NPIX/16), 32, 0, stream>>>(
        at16, 64, wp16, 64, nat_proj_b + i*64,
        x + i*64, 256, x1, 64, nullptr, 0, 64);

    // LN2 -> f16
    k_ln64<true, false><<<lnGrid, 128, 0, stream>>>(x1, 64,
        nat_ln2_g + i*64, nat_ln2_b + i*64, ln216, nullptr, NPIX);

    // fc1 + bias + GELU -> f16
    k_gemm<1, false, false, true><<<dim3(256/16, NPIX/16), 32, 0, stream>>>(
        ln216, 64, w116, 64, nat_fc1_b + i*256,
        nullptr, 0, nullptr, 0, h116, 256, 64);

    // fc2 + bias + residual(x1) -> d_out (strided, channel offset i*64)
    k_gemm<0, true, true, false><<<dim3(64/16, NPIX/16), 32, 0, stream>>>(
        h116, 256, w216, 256, nat_fc2_b + i*64,
        x1, 64, out + i*64, 256, nullptr, 0, 256);
  }

  // ---- global attention branch (channels 192..255) ----
  k_cvtT16<<<(64*192 + 255) / 256, 256, 0, stream>>>(ga_qkv_w, wq16, 64, 192);
  k_cvtT16<<<(64*192 + 255) / 256, 256, 0, stream>>>(ga_fc1_w, w116, 64, 192);
  k_cvtT16<<<(192*64 + 255) / 256, 256, 0, stream>>>(ga_fc2_w, w216, 192, 64);

  k_ln64<false, true><<<lnGrid, 128, 0, stream>>>(x + 192, 256,
      ga_ln1_g, ga_ln1_b, nullptr, xn32, NPIX);
  k_ga_pool<<<(NPOOL*64 + 255) / 256, 256, 0, stream>>>(xn32, xn16);

  k_gemm<0, false, true, false><<<dim3(192/16, NPOOL/16), 32, 0, stream>>>(
      xn16, 64, wq16, 64, ga_qkv_b,
      nullptr, 0, qkv32, 192, nullptr, 0, 64);

  float* ga_ao = (float*)at16;   // 3136*64 f32 fits in the at16 region
  k_ga_attn<<<(NPOOL*4 + 255) / 256, 256, 0, stream>>>(qkv32, ga_ao);
  k_ga_upadd<<<(NPIX*64 + 255) / 256, 256, 0, stream>>>(x, ga_ao, x1);

  k_ln64<true, false><<<lnGrid, 128, 0, stream>>>(x1, 64,
      ga_ln2_g, ga_ln2_b, ln216, nullptr, NPIX);

  k_gemm<1, false, false, true><<<dim3(192/16, NPIX/16), 32, 0, stream>>>(
      ln216, 64, w116, 64, ga_fc1_b,
      nullptr, 0, nullptr, 0, h116, 192, 64);

  k_gemm<0, true, true, false><<<dim3(64/16, NPIX/16), 32, 0, stream>>>(
      h116, 192, w216, 192, ga_fc2_b,
      x1, 64, out + 192, 256, nullptr, 0, 192);
}
